// TransformerXL_67680094650929
// MI455X (gfx1250) — matrix operand order repaired and checked by reading, not compiled, not run on hardware
//
#include <hip/hip_runtime.h>

// ---------------- problem constants ----------------
#define L_    4
#define B_    8
#define T_    512
#define M_    512
#define E_    512
#define H_    8
#define D_    64
#define F_    2048
#define KLEN_ 1024
#define HD_   512

typedef _Float16 h16;
typedef _Float16 v16h __attribute__((ext_vector_type(16)));
typedef _Float16 v8h  __attribute__((ext_vector_type(8)));
typedef float    v8f  __attribute__((ext_vector_type(8)));
typedef int      b128i __attribute__((vector_size(16)));

// ---------------- CDNA5 async-to-LDS helpers (guarded) ----------------
#if defined(__HIP_DEVICE_COMPILE__) && __has_builtin(__builtin_amdgcn_s_wait_asynccnt)
#define WAIT_ASYNC(n) __builtin_amdgcn_s_wait_asynccnt(n)
#elif defined(__HIP_DEVICE_COMPILE__)
#define WAIT_ASYNC(n) asm volatile("s_wait_asynccnt %0" ::"i"(n) : "memory")
#else
#define WAIT_ASYNC(n) ((void)0)
#endif

__device__ __forceinline__ void async_b128(void* g, void* l) {
#if defined(__HIP_DEVICE_COMPILE__) && __has_builtin(__builtin_amdgcn_global_load_async_to_lds_b128)
  typedef __attribute__((address_space(1))) b128i* gp_t;
  typedef __attribute__((address_space(3))) b128i* lp_t;
  __builtin_amdgcn_global_load_async_to_lds_b128((gp_t)g, (lp_t)l, 0, 0);
#else
  *(v8h*)l = *(const v8h*)g;
#endif
}

// ---------------- batched WMMA GEMM: C[M,N] = A[M,K] * W[N,K]^T ----------------
// A,W f16 row-major (K contiguous). C f32. Optional bias over N, ReLU.
// Batch z: A += z*sA, W += (z%bmod)*sB, C += z*sC.
__global__ __launch_bounds__(256) void k_wmma_gemm(
    const h16* __restrict__ A, const h16* __restrict__ Bw, float* __restrict__ C,
    const float* __restrict__ bias, int Md, int Nd, int Kd,
    long long sA, long long sB, long long sC, int bmod, float alpha, int relu)
{
  __shared__ __align__(16) h16 As[2][128][32];
  __shared__ __align__(16) h16 Bs[2][128][32];

  const int bz = blockIdx.z;
  const h16* Ab = A + (long long)bz * sA;
  const h16* Bb = Bw + (long long)(bz % bmod) * sB;
  float* Cb = C + (long long)bz * sC;

  const int blockM = blockIdx.y * 128;
  const int blockN = blockIdx.x * 128;
  const int t = threadIdx.x;
  const int lane = t & 31;
  const int w = t >> 5;               // 8 waves
  const int wm = (w >> 2) * 64;       // wave M origin: 0 or 64
  const int wn = (w & 3) * 32;        // wave N origin: 0,32,64,96
  const int l15 = lane & 15;
  const int hi = lane >> 4;           // half-wave select

  const bool fullA = (blockM + 128 <= Md);
  const bool fullB = (blockN + 128 <= Nd);

  v8f acc[4][2];
#pragma unroll
  for (int mt = 0; mt < 4; ++mt)
#pragma unroll
    for (int nt = 0; nt < 2; ++nt)
#pragma unroll
      for (int q = 0; q < 8; ++q) acc[mt][nt][q] = 0.f;

  auto load_stage = [&](int kb, int buf) {
    const int kk = kb * 32;
#pragma unroll
    for (int c = 0; c < 2; ++c) {
      const int chunk = (t << 1) | c;      // 0..511, 8 halves each
      const int row = chunk >> 2;          // 0..127
      const int col = (chunk & 3) << 3;    // 0,8,16,24
      {  // A tile
        const int gr = blockM + row;
        const h16* gp = Ab + (long long)gr * Kd + (kk + col);
        h16* lp = &As[buf][row][col];
        if (fullA) {
          async_b128((void*)gp, lp);
        } else {
          v8h v;
#pragma unroll
          for (int q = 0; q < 8; ++q) v[q] = (h16)0.f;
          if (gr < Md) v = *(const v8h*)gp;
          *(v8h*)lp = v;
        }
      }
      {  // B tile
        const int gr = blockN + row;
        const h16* gp = Bb + (long long)gr * Kd + (kk + col);
        h16* lp = &Bs[buf][row][col];
        if (fullB) {
          async_b128((void*)gp, lp);
        } else {
          v8h v;
#pragma unroll
          for (int q = 0; q < 8; ++q) v[q] = (h16)0.f;
          if (gr < Nd) v = *(const v8h*)gp;
          *(v8h*)lp = v;
        }
      }
    }
  };

  load_stage(0, 0);
  WAIT_ASYNC(0);
  __syncthreads();

  const int nK = Kd >> 5;
  for (int kb = 0; kb < nK; ++kb) {
    const int buf = kb & 1;
    if (kb + 1 < nK) load_stage(kb + 1, buf ^ 1);

    // A fragments: 16x32 f16, lanes0-15 K{0..7,16..23}, lanes16-31 K{8..15,24..31}
    v16h af[4];
#pragma unroll
    for (int mt = 0; mt < 4; ++mt) {
      const int row = wm + mt * 16 + l15;
      const int kb0 = hi << 3;
      v8h lo = *(const v8h*)&As[buf][row][kb0];
      v8h hh = *(const v8h*)&As[buf][row][kb0 + 16];
#pragma unroll
      for (int q = 0; q < 8; ++q) { af[mt][q] = lo[q]; af[mt][q + 8] = hh[q]; }
    }
    // B fragments: 32x16, lanes0-15 K{0..7,16..23}, lanes16-31 K{8..15,24..31} (N = lane&15)
    v16h bf2[2];
#pragma unroll
    for (int nt = 0; nt < 2; ++nt) {
      const int colr = wn + nt * 16 + l15;
      const int koff = hi << 3;
      v8h lo = *(const v8h*)&Bs[buf][colr][koff];
      v8h hh = *(const v8h*)&Bs[buf][colr][koff + 16];
#pragma unroll
      for (int q = 0; q < 8; ++q) { bf2[nt][q] = lo[q]; bf2[nt][q + 8] = hh[q]; }
    }

#pragma unroll
    for (int mt = 0; mt < 4; ++mt)
#pragma unroll
      for (int nt = 0; nt < 2; ++nt)
        acc[mt][nt] = __builtin_amdgcn_wmma_f32_16x16x32_f16(
            false, af[mt], false, bf2[nt], (short)0, acc[mt][nt], false, false);

    WAIT_ASYNC(0);
    __syncthreads();
  }

  // Epilogue: C/D layout -> VGPR r holds M = r + 8*hi, N = lane&15
#pragma unroll
  for (int mt = 0; mt < 4; ++mt)
#pragma unroll
    for (int nt = 0; nt < 2; ++nt)
#pragma unroll
      for (int r = 0; r < 8; ++r) {
        const int gm = blockM + wm + mt * 16 + r + (hi << 3);
        const int gn = blockN + wn + nt * 16 + l15;
        if (gm < Md && gn < Nd) {
          float v = acc[mt][nt][r] * alpha;
          if (bias) v += bias[gn];
          if (relu) v = fmaxf(v, 0.f);
          Cb[(long long)gm * Nd + gn] = v;
        }
      }
}

// ---------------- elementwise / support kernels ----------------
__global__ void k_cvt(const float* __restrict__ in, h16* __restrict__ out, long long n) {
  for (long long i = (long long)blockIdx.x * blockDim.x + threadIdx.x; i < n;
       i += (long long)gridDim.x * blockDim.x)
    out[i] = (h16)in[i];
}

// x [B,T,E] -> h [T,B,E]
__global__ void k_transpose_in(const float* __restrict__ x, float* __restrict__ h, long long n) {
  for (long long i = (long long)blockIdx.x * blockDim.x + threadIdx.x; i < n;
       i += (long long)gridDim.x * blockDim.x) {
    const int e = (int)(i % E_);
    long long tmp = i / E_;
    const int b = (int)(tmp % B_);
    const long long tt = tmp / B_;
    h[i] = x[((long long)b * T_ + tt) * E_ + e];
  }
}

__global__ void k_copy(const float* __restrict__ in, float* __restrict__ out, long long n) {
  for (long long i = (long long)blockIdx.x * blockDim.x + threadIdx.x; i < n;
       i += (long long)gridDim.x * blockDim.x)
    out[i] = in[i];
}

// sinusoid table, f32 + f16
__global__ __launch_bounds__(256) void k_pe(const int* __restrict__ positions,
                                            float* __restrict__ pe, h16* __restrict__ peh) {
  const int k = blockIdx.x;
  const float p = (float)positions[k];
#pragma unroll
  for (int c = 0; c < 2; ++c) {
    const int e = threadIdx.x + c * 256;
    const int f = (e < 256) ? e : e - 256;
    const float invf = __expf(-9.2103403720f * (float)(2 * f) / (float)E_);
    const float ang = p * invf;
    const float v = (e < 256) ? __sinf(ang) : __cosf(ang);
    pe[(long long)k * E_ + e] = v;
    peh[(long long)k * E_ + e] = (h16)v;
  }
}

// c[k*B+b, e] = (k < M) ? mems_i : h, as f16
__global__ void k_concat(const float* __restrict__ memsI, const float* __restrict__ h,
                         h16* __restrict__ ch, long long n) {
  for (long long i = (long long)blockIdx.x * blockDim.x + threadIdx.x; i < n;
       i += (long long)gridDim.x * blockDim.x) {
    const int e = (int)(i % E_);
    long long tmp = i / E_;
    const int b = (int)(tmp % B_);
    const long long k = tmp / B_;
    float v;
    if (k < M_) v = memsI[((long long)k * B_ + b) * E_ + e];
    else        v = h[((k - M_) * B_ + b) * E_ + e];
    ch[i] = (h16)v;
  }
}

// qkv [KLEN*B, 3*HD] -> qu/qv [bh,T,D] f16 (+biases), k [bh,KLEN,D], vT [bh,D,KLEN]
__global__ void k_repack_qkv(const float* __restrict__ qkv, const float* __restrict__ ub,
                             const float* __restrict__ vb, h16* __restrict__ quh,
                             h16* __restrict__ qvh, h16* __restrict__ kh,
                             h16* __restrict__ vth, long long n) {
  for (long long i = (long long)blockIdx.x * blockDim.x + threadIdx.x; i < n;
       i += (long long)gridDim.x * blockDim.x) {
    const int d = (int)(i % D_);
    long long tmp = i / D_;
    const int jk = (int)(tmp % KLEN_);
    tmp /= KLEN_;
    const int hh = (int)(tmp % H_);
    const int b = (int)(tmp / H_);
    const long long row = (long long)jk * B_ + b;
    const float* qr = qkv + row * (3 * HD_);
    const long long bh = (long long)b * H_ + hh;
    kh[(bh * KLEN_ + jk) * D_ + d]  = (h16)qr[HD_ + hh * D_ + d];
    vth[(bh * D_ + d) * KLEN_ + jk] = (h16)qr[2 * HD_ + hh * D_ + d];
    if (jk >= M_) {
      const int ii = jk - M_;
      const float q = qr[hh * D_ + d];
      quh[(bh * T_ + ii) * D_ + d] = (h16)(q + ub[hh * D_ + d]);
      qvh[(bh * T_ + ii) * D_ + d] = (h16)(q + vb[hh * D_ + d]);
    }
  }
}

// r [KLEN, HD] f32 -> [H, KLEN, D] f16
__global__ void k_repack_r(const float* __restrict__ rf, h16* __restrict__ rh, long long n) {
  for (long long i = (long long)blockIdx.x * blockDim.x + threadIdx.x; i < n;
       i += (long long)gridDim.x * blockDim.x) {
    const int d = (int)(i % D_);
    long long tmp = i / D_;
    const int jk = (int)(tmp % KLEN_);
    const int hh = (int)(tmp / KLEN_);
    rh[((long long)hh * KLEN_ + jk) * D_ + d] = (h16)rf[(long long)jk * HD_ + hh * D_ + d];
  }
}

// O [B,H,T,D] f32 -> o [T*B, HD] f16
__global__ void k_repack_o(const float* __restrict__ Of, h16* __restrict__ oh, long long n) {
  for (long long i = (long long)blockIdx.x * blockDim.x + threadIdx.x; i < n;
       i += (long long)gridDim.x * blockDim.x) {
    const int col = (int)(i % HD_);
    long long tmp = i / HD_;
    const int b = (int)(tmp % B_);
    const long long ii = tmp / B_;
    const int hh = col / D_;
    const int d = col % D_;
    oh[i] = (h16)Of[(((long long)b * H_ + hh) * T_ + ii) * D_ + d];
  }
}

// fused TXL rel_shift + mask + softmax; writes f16 probabilities
__global__ __launch_bounds__(256) void k_softmax(const float* __restrict__ Sc,
                                                 const float* __restrict__ Sp,
                                                 h16* __restrict__ attn, float scale) {
  const int i = blockIdx.x;
  const int hh = blockIdx.y;
  const int b = blockIdx.z;
  const long long bh = (long long)b * H_ + hh;
  const float* sc = Sc + (bh * T_ + i) * (long long)KLEN_;
  const float* sp = Sp + bh * (long long)T_ * KLEN_;
  h16* ao = attn + (bh * T_ + i) * (long long)KLEN_;
  const int t = threadIdx.x;
  __shared__ float red[8];
  float vals[4];
  float mx = -3e38f;
#pragma unroll
  for (int c = 0; c < 4; ++c) {
    const int j = t + c * 256;
    // rel_shift gather: out[i,j] = pad-reshape trick over [T, KLEN+1]
    const long long idx = (long long)T_ + (long long)i * KLEN_ + j;
    const int i0 = (int)(idx / (KLEN_ + 1));
    const int j0 = (int)(idx % (KLEN_ + 1));
    float pv = 0.f;
    if (j0 != 0 && i0 < T_) pv = sp[(long long)i0 * KLEN_ + (j0 - 1)];
    float v = (sc[j] + pv) * scale;
    if (j > i + M_) v = -1e30f;
    vals[c] = v;
    mx = fmaxf(mx, v);
  }
  for (int o = 16; o; o >>= 1) mx = fmaxf(mx, __shfl_xor(mx, o, 32));
  if ((t & 31) == 0) red[t >> 5] = mx;
  __syncthreads();
  float gm = red[0];
#pragma unroll
  for (int q = 1; q < 8; ++q) gm = fmaxf(gm, red[q]);
  __syncthreads();
  float s = 0.f;
#pragma unroll
  for (int c = 0; c < 4; ++c) { vals[c] = __expf(vals[c] - gm); s += vals[c]; }
  for (int o = 16; o; o >>= 1) s += __shfl_xor(s, o, 32);
  if ((t & 31) == 0) red[t >> 5] = s;
  __syncthreads();
  float gs = 0.f;
#pragma unroll
  for (int q = 0; q < 8; ++q) gs += red[q];
  const float inv = 1.f / gs;
#pragma unroll
  for (int c = 0; c < 4; ++c) ao[t + c * 256] = (h16)(vals[c] * inv);
}

// out = LN(add + res); writes f32 and f16. One row (E=512) per block.
__global__ __launch_bounds__(256) void k_ln(const float* __restrict__ add,
                                            const float* __restrict__ res,
                                            const float* __restrict__ g,
                                            const float* __restrict__ bta,
                                            float* __restrict__ outf, h16* __restrict__ outh) {
  const long long row = blockIdx.x;
  const float* a = add + row * E_;
  const float* r = res + row * E_;
  const int t = threadIdx.x;
  const float v0 = a[t] + r[t];
  const float v1 = a[t + 256] + r[t + 256];
  __shared__ float red[8];
  float s = v0 + v1;
  for (int o = 16; o; o >>= 1) s += __shfl_xor(s, o, 32);
  if ((t & 31) == 0) red[t >> 5] = s;
  __syncthreads();
  float mean = 0.f;
#pragma unroll
  for (int q = 0; q < 8; ++q) mean += red[q];
  mean *= (1.f / E_);
  __syncthreads();
  const float d0 = v0 - mean, d1 = v1 - mean;
  float sq = d0 * d0 + d1 * d1;
  for (int o = 16; o; o >>= 1) sq += __shfl_xor(sq, o, 32);
  if ((t & 31) == 0) red[t >> 5] = sq;
  __syncthreads();
  float var = 0.f;
#pragma unroll
  for (int q = 0; q < 8; ++q) var += red[q];
  var *= (1.f / E_);
  const float rs = rsqrtf(var + 1e-5f);
  const float o0 = d0 * rs * g[t] + bta[t];
  const float o1 = d1 * rs * g[t + 256] + bta[t + 256];
  float* of = outf + row * E_;
  h16* oh = outh + row * E_;
  of[t] = o0; of[t + 256] = o1;
  oh[t] = (h16)o0; oh[t + 256] = (h16)o1;
}

// final: out [B,T,E] then new_mems [(L+1),M,B,E]
__global__ void k_writeout(const float* __restrict__ h, const float* __restrict__ mems,
                           const float* __restrict__ hid0, float* __restrict__ out,
                           long long n) {
  const long long nBTE = (long long)B_ * T_ * E_;
  for (long long i = (long long)blockIdx.x * blockDim.x + threadIdx.x; i < n;
       i += (long long)gridDim.x * blockDim.x) {
    if (i < nBTE) {
      const int e = (int)(i % E_);
      long long tmp = i / E_;
      const int tt = (int)(tmp % T_);
      const int b = (int)(tmp / T_);
      out[i] = h[((long long)tt * B_ + b) * E_ + e];
    } else {
      const long long j = i - nBTE;
      const int e = (int)(j % E_);
      long long tmp = j / E_;
      const int b = (int)(tmp % B_);
      tmp /= B_;
      const int m = (int)(tmp % M_);
      const int li = (int)(tmp / M_);
      float v;
      if (m < M_ - 1)
        v = mems[(((long long)li * M_ + (m + 1)) * B_ + b) * E_ + e];
      else
        v = hid0[(long long)li * B_ * E_ + (long long)b * E_ + e];
      out[i] = v;
    }
  }
}

// ---------------- host orchestration ----------------
extern "C" void kernel_launch(void* const* d_in, const int* in_sizes, int n_in,
                              void* d_out, int out_size, void* d_ws, size_t ws_size,
                              hipStream_t stream) {
  (void)in_sizes; (void)n_in; (void)out_size; (void)ws_size;
  const float* x      = (const float*)d_in[0];
  const float* mems   = (const float*)d_in[1];
  const int*   pos    = (const int*)d_in[2];
  /* d_in[3] attn_mask: computed analytically */
  const float* u_bias = (const float*)d_in[4];
  const float* v_bias = (const float*)d_in[5];
  const float* qkv_w  = (const float*)d_in[6];
  const float* pos_w  = (const float*)d_in[7];
  const float* out_w  = (const float*)d_in[8];
  const float* out_b  = (const float*)d_in[9];
  const float* ff1_w  = (const float*)d_in[10];
  const float* ff1_b  = (const float*)d_in[11];
  const float* ff2_w  = (const float*)d_in[12];
  const float* ff2_b  = (const float*)d_in[13];
  const float* ln1_g  = (const float*)d_in[14];
  const float* ln1_b  = (const float*)d_in[15];
  const float* ln2_g  = (const float*)d_in[16];
  const float* ln2_b  = (const float*)d_in[17];

  char* base = (char*)d_ws;
  size_t off = 0;
  auto alloc = [&](size_t bytes) -> void* {
    void* p = base + off;
    off = (off + bytes + 255) & ~(size_t)255;
    return p;
  };

  const size_t nTBE = (size_t)T_ * B_ * E_;
  float* h_f   = (float*)alloc(nTBE * 4);
  h16*   h_h   = (h16*)alloc(nTBE * 2);
  float* hid0  = (float*)alloc((size_t)(L_ + 1) * B_ * E_ * 4);
  float* pe_f  = (float*)alloc((size_t)KLEN_ * E_ * 4);
  h16*   pe_h  = (h16*)alloc((size_t)KLEN_ * E_ * 2);
  h16* qkvw_h  = (h16*)alloc((size_t)L_ * 3 * HD_ * E_ * 2);
  h16* posw_h  = (h16*)alloc((size_t)L_ * HD_ * E_ * 2);
  h16* outw_h  = (h16*)alloc((size_t)L_ * E_ * HD_ * 2);
  h16* ff1w_h  = (h16*)alloc((size_t)L_ * F_ * E_ * 2);
  h16* ff2w_h  = (h16*)alloc((size_t)L_ * E_ * F_ * 2);
  h16*   c_h   = (h16*)alloc((size_t)KLEN_ * B_ * E_ * 2);
  float* qkv_f = (float*)alloc((size_t)KLEN_ * B_ * 3 * HD_ * 4);
  float* r_f   = (float*)alloc((size_t)KLEN_ * HD_ * 4);
  h16*   r_h   = (h16*)alloc((size_t)H_ * KLEN_ * D_ * 2);
  h16*   qu_h  = (h16*)alloc((size_t)B_ * H_ * T_ * D_ * 2);
  h16*   qv_h  = (h16*)alloc((size_t)B_ * H_ * T_ * D_ * 2);
  h16*   kk_h  = (h16*)alloc((size_t)B_ * H_ * KLEN_ * D_ * 2);
  h16*   vt_h  = (h16*)alloc((size_t)B_ * H_ * KLEN_ * D_ * 2);
  float* Sc    = (float*)alloc((size_t)B_ * H_ * T_ * KLEN_ * 4);
  float* Sp    = (float*)alloc((size_t)B_ * H_ * T_ * KLEN_ * 4);
  h16*   at_h  = (h16*)alloc((size_t)B_ * H_ * T_ * KLEN_ * 2);
  float* O_f   = (float*)alloc((size_t)B_ * H_ * T_ * D_ * 4);
  h16*   o_h   = (h16*)alloc(nTBE * 2);
  float* aout  = (float*)alloc(nTBE * 4);
  float* ffh_f = (float*)alloc((size_t)T_ * B_ * F_ * 4);
  h16*   ffh_h = (h16*)alloc((size_t)T_ * B_ * F_ * 2);
  float* ff2o  = (float*)alloc(nTBE * 4);

  auto eg = [](long long n) -> unsigned {
    long long g = (n + 255) / 256;
    return (unsigned)(g > 16384 ? 16384 : g);
  };
  auto gemm = [&](const h16* Aq, const h16* Bq, float* Cq, const float* bias, int Md,
                  int Nd, int Kd, int batch, long long sAq, long long sBq, long long sCq,
                  int bmod, float alpha, int relu) {
    dim3 grid((Nd + 127) / 128, (Md + 127) / 128, batch);
    k_wmma_gemm<<<grid, 256, 0, stream>>>(Aq, Bq, Cq, bias, Md, Nd, Kd, sAq, sBq, sCq,
                                          bmod, alpha, relu);
  };

  // --- weights f32 -> f16 (L2 resident thereafter) ---
  k_cvt<<<eg((long long)L_ * 3 * HD_ * E_), 256, 0, stream>>>(qkv_w, qkvw_h, (long long)L_ * 3 * HD_ * E_);
  k_cvt<<<eg((long long)L_ * HD_ * E_), 256, 0, stream>>>(pos_w, posw_h, (long long)L_ * HD_ * E_);
  k_cvt<<<eg((long long)L_ * E_ * HD_), 256, 0, stream>>>(out_w, outw_h, (long long)L_ * E_ * HD_);
  k_cvt<<<eg((long long)L_ * F_ * E_), 256, 0, stream>>>(ff1_w, ff1w_h, (long long)L_ * F_ * E_);
  k_cvt<<<eg((long long)L_ * E_ * F_), 256, 0, stream>>>(ff2_w, ff2w_h, (long long)L_ * E_ * F_);

  // --- h = transpose(x), pe table, hiddens[0][0] ---
  k_transpose_in<<<eg((long long)nTBE), 256, 0, stream>>>(x, h_f, (long long)nTBE);
  k_pe<<<KLEN_, 256, 0, stream>>>(pos, pe_f, pe_h);
  k_copy<<<eg((long long)B_ * E_), 256, 0, stream>>>(h_f, hid0, (long long)B_ * E_);

  const float scale = 0.125f;  // 1/sqrt(64)

  for (int i = 0; i < L_; ++i) {
    const float* memsI = mems + (size_t)i * M_ * B_ * E_;
    // c = concat(mems[i], h) as f16
    k_concat<<<eg((long long)KLEN_ * B_ * E_), 256, 0, stream>>>(memsI, h_f, c_h,
                                                                 (long long)KLEN_ * B_ * E_);
    // qkv = c @ qkv_w^T    [8192 x 1536]
    gemm(c_h, qkvw_h + (size_t)i * 3 * HD_ * E_, qkv_f, nullptr, KLEN_ * B_, 3 * HD_, E_,
         1, 0, 0, 0, 1, 1.f, 0);
    // r = pe @ pos_w^T     [1024 x 512]
    gemm(pe_h, posw_h + (size_t)i * HD_ * E_, r_f, nullptr, KLEN_, HD_, E_, 1, 0, 0, 0, 1,
         1.f, 0);
    k_repack_r<<<eg((long long)H_ * KLEN_ * D_), 256, 0, stream>>>(r_f, r_h,
                                                                   (long long)H_ * KLEN_ * D_);
    k_repack_qkv<<<eg((long long)B_ * H_ * KLEN_ * D_), 256, 0, stream>>>(
        qkv_f, u_bias, v_bias, qu_h, qv_h, kk_h, vt_h, (long long)B_ * H_ * KLEN_ * D_);
    // content scores: per (b,h)  [T x KLEN], K=64
    gemm(qu_h, kk_h, Sc, nullptr, T_, KLEN_, D_, B_ * H_, (long long)T_ * D_,
         (long long)KLEN_ * D_, (long long)T_ * KLEN_, B_ * H_, 1.f, 0);
    // position scores: B-operand broadcasts over batch (z % H)
    gemm(qv_h, r_h, Sp, nullptr, T_, KLEN_, D_, B_ * H_, (long long)T_ * D_,
         (long long)KLEN_ * D_, (long long)T_ * KLEN_, H_, 1.f, 0);
    // softmax (rel_shift + causal-mem mask fused)
    k_softmax<<<dim3(T_, H_, B_), 256, 0, stream>>>(Sc, Sp, at_h, scale);
    // O = P @ V  (V pre-transposed to [D, KLEN])
    gemm(at_h, vt_h, O_f, nullptr, T_, D_, KLEN_, B_ * H_, (long long)T_ * KLEN_,
         (long long)D_ * KLEN_, (long long)T_ * D_, B_ * H_, 1.f, 0);
    k_repack_o<<<eg((long long)nTBE), 256, 0, stream>>>(O_f, o_h, (long long)nTBE);
    // out projection + bias
    gemm(o_h, outw_h + (size_t)i * E_ * HD_, aout, out_b + (size_t)i * E_, T_ * B_, E_,
         HD_, 1, 0, 0, 0, 1, 1.f, 0);
    // h = LN(h + attn_out)
    k_ln<<<T_ * B_, 256, 0, stream>>>(aout, h_f, ln1_g + (size_t)i * E_,
                                      ln1_b + (size_t)i * E_, h_f, h_h);
    // FF1 (+bias, ReLU)
    gemm(h_h, ff1w_h + (size_t)i * F_ * E_, ffh_f, ff1_b + (size_t)i * F_, T_ * B_, F_,
         E_, 1, 0, 0, 0, 1, 1.f, 1);
    k_cvt<<<eg((long long)T_ * B_ * F_), 256, 0, stream>>>(ffh_f, ffh_h,
                                                           (long long)T_ * B_ * F_);
    // FF2 (+bias)
    gemm(ffh_h, ff2w_h + (size_t)i * E_ * F_, ff2o, ff2_b + (size_t)i * E_, T_ * B_, E_,
         F_, 1, 0, 0, 0, 1, 1.f, 0);
    // h = LN(h + ff)
    k_ln<<<T_ * B_, 256, 0, stream>>>(ff2o, h_f, ln2_g + (size_t)i * E_,
                                      ln2_b + (size_t)i * E_, h_f, h_h);
    // hiddens[i+1][0]
    k_copy<<<eg((long long)B_ * E_), 256, 0, stream>>>(h_f, hid0 + (size_t)(i + 1) * B_ * E_,
                                                       (long long)B_ * E_);
  }

  const long long total = (long long)B_ * T_ * E_ + (long long)(L_ + 1) * M_ * B_ * E_;
  k_writeout<<<eg(total), 256, 0, stream>>>(h_f, mems, hid0, (float*)d_out, total);
}
